// BEVConvSNormal_19696720019811
// MI455X (gfx1250) — compile-verified, matrix operand order checked by reading, not run on hardware
//
#include <hip/hip_runtime.h>

// ---------- CDNA5 WMMA types ----------
typedef __attribute__((ext_vector_type(16))) __bf16 v16bf;
typedef __attribute__((ext_vector_type(8)))  float  v8f;

// ---------- helpers ----------
__device__ inline unsigned map_f(float f) {
  union { float f; unsigned u; } v; v.f = f;
  return (v.u & 0x80000000u) ? ~v.u : (v.u | 0x80000000u);
}
__device__ inline float unmap_f(unsigned m) {
  unsigned u = (m & 0x80000000u) ? (m & 0x7fffffffu) : ~m;
  union { unsigned u; float f; } v; v.u = u; return v.f;
}
__device__ inline float u2f(unsigned u) {
  union { unsigned u; float f; } v; v.u = u; return v.f;
}
__device__ inline __bf16 f2bf(float f) {
  union { float f; unsigned u; } v; v.f = f;
  unsigned u = v.u + (0x7fffu + ((v.u >> 16) & 1u));   // round-to-nearest-even
  unsigned short h = (unsigned short)(u >> 16);
  __bf16 b; __builtin_memcpy(&b, &h, 2);
  return b;
}

// ---------- 1) init: bev buffers (monotone-mapped u32) + BN stats ----------
__global__ void init_kernel(unsigned* __restrict__ ubev, float* __restrict__ stats,
                            long long nbev) {
  long long i = (long long)blockIdx.x * blockDim.x + threadIdx.x;
  if (i < 768) stats[i] = 0.0f;                 // stats + scale + shift scratch
  if (i >= nbev) return;
  int c = (int)((i >> 20) & 1);                 // layout [b][c][1024][1024]
  ubev[i] = (c == 0) ? map_f(-10.0f) : map_f(0.0f);
}

// ---------- 2) point scatter with integer atomicMax on mapped floats ----------
__global__ void scatter_kernel(const float* __restrict__ pts,
                               unsigned* __restrict__ ubev, int N) {
  int i = blockIdx.x * blockDim.x + threadIdx.x;
  if (i >= N) return;
  float bf = pts[i * 5 + 0];
  float x  = pts[i * 5 + 1];
  float y  = pts[i * 5 + 2];
  float z  = pts[i * 5 + 3];
  float it = pts[i * 5 + 4];
  const float xs = 1024.0f / 69.12f;
  const float ys = 1024.0f / 79.36f;
  int xp = (int)(x * xs);                       // trunc, matches astype(int32)
  int yp = (int)((y + 39.68f) * ys);
  if (xp < 0 || xp >= 1024 || yp < 0 || yp >= 1024) return;
  int b = (int)bf;
  unsigned idx0 = (((unsigned)b * 2u) << 20) + ((unsigned)yp << 10) + (unsigned)xp;
  unsigned idx1 = idx0 + (1u << 20);
  atomicMax(&ubev[idx0], map_f(z));
  atomicMax(&ubev[idx1], map_f(it));
}

// ---------- 3) conv3x3 (SAME) as implicit GEMM on v_wmma_f32_16x16x32_bf16 ----
// One block = 256 threads (8 waves), one (b, y, 128-pixel x span).
// Each wave: 16(co) x 16(pixel) tile, K = CIN*9 padded to multiple of 32.
// Input patch is staged global->LDS with CDNA5 async-to-LDS DMA (ASYNCcnt),
// then converted to bf16 in one LDS pass. Per-channel BN statistics are
// accumulated in the epilogue (LDS reduction + one global atomicAdd).
template<int CIN, int COUT, bool MAP>
__global__ __launch_bounds__(256)
void conv_wmma_kernel(const void* __restrict__ in_raw,
                      const float* __restrict__ wght,
                      const float* __restrict__ bias,
                      float* __restrict__ out,
                      float* __restrict__ stats,
                      int H, int W) {
  constexpr int K    = CIN * 9;
  constexpr int KP   = (K + 31) & ~31;
  constexpr int NCO  = (COUT + 15) / 16;
  constexpr int PELT = CIN * 3 * 130;           // valid patch elements

  __shared__ __bf16   wlds[16 * NCO * KP];      // [co][k], zero-padded
  __shared__ unsigned ustage[CIN * 3 * 132];    // raw f32 / mapped-u32 patch
  __shared__ __bf16   patch[CIN * 3 * 132];     // bf16 patch, stride 132
  __shared__ float    sstat[COUT * 2];

  const int tid  = threadIdx.x;
  const int wave = tid >> 5;
  const int lane = tid & 31;
  const int x0   = blockIdx.x * 128;
  const int y    = blockIdx.y;
  const int b    = blockIdx.z;

  for (int i = tid; i < COUT * 2; i += 256) sstat[i] = 0.0f;

  // stage weights (tiny, L2-resident): [co][k] bf16, zero-padded
  for (int i = tid; i < 16 * NCO * KP; i += 256) {
    int co = i / KP, k = i % KP;
    float v = (co < COUT && k < K) ? wght[co * K + k] : 0.0f;
    wlds[i] = f2bf(v);
  }

  // stage input patch rows y-1..y+1, x0-1..x0+128 via async global->LDS DMA.
  // In-bounds elements: GLOBAL_LOAD_ASYNC_TO_LDS_B32 (no VGPR return, ASYNCcnt).
  // OOB/padding elements: plain ds_store of the zero pattern (disjoint addrs,
  // so no DS-vs-ASYNC ordering hazard).
  const unsigned* inu = (const unsigned*)in_raw;
  const unsigned zpat = MAP ? map_f(0.0f) : 0u;
  for (int i = tid; i < PELT; i += 256) {
    int ci = i / (3 * 130);
    int r  = (i / 130) % 3;
    int xx = i % 130;
    int gy = y + r - 1;
    int gx = x0 + xx - 1;
    unsigned* sp = &ustage[(ci * 3 + r) * 132 + xx];
    if (gy >= 0 && gy < H && gx >= 0 && gx < W) {
      const unsigned* gp = &inu[(((long long)b * CIN + ci) * H + gy) * W + gx];
      unsigned lds_off = (unsigned)(uintptr_t)sp;   // low 32 bits = LDS offset
      asm volatile("global_load_async_to_lds_b32 %0, %1, off"
                   :: "v"(lds_off), "v"(gp) : "memory");
    } else {
      *sp = zpat;
    }
  }
  asm volatile("s_wait_asynccnt 0x0" ::: "memory");
  __syncthreads();

  // convert staged raw words -> bf16 patch (decodes conv1's monotone mapping)
  for (int i = tid; i < PELT; i += 256) {
    int ci = i / (3 * 130);
    int r  = (i / 130) % 3;
    int xx = i % 130;
    int idx = (ci * 3 + r) * 132 + xx;
    unsigned u = ustage[idx];
    float v = MAP ? unmap_f(u) : u2f(u);
    patch[idx] = f2bf(v);
  }
  __syncthreads();

  const int half = lane >> 4;
  const int ln   = lane & 15;
  const int px   = wave * 16 + ln;              // local pixel (B-fragment N, output x)

  for (int ct = 0; ct < NCO; ++ct) {
    v8f c = {};
#pragma unroll
    for (int k0 = 0; k0 < KP; k0 += 32) {
      v16bf a, bm;
#pragma unroll
      for (int j = 0; j < 16; ++j) {
        // 16-bit fragment K layout (ISA 7.12.2): K = base + (j&8)*2 + half*8 + (j&7)
        int k = k0 + ((j & 8) << 1) + (half << 3) + (j & 7);
        a[j] = wlds[(ct * 16 + ln) * KP + k];   // A: M = ln (co row)
        __bf16 bv;
        if (k < K) {
          int ci = k / 9, rem = k % 9;
          int dy = rem / 3, dx = rem % 3;
          bv = patch[(ci * 3 + dy) * 132 + (px + dx)];  // B: N = ln (pixel)
        } else {
          bv = f2bf(0.0f);
        }
        bm[j] = bv;
      }
      c = __builtin_amdgcn_wmma_f32_16x16x32_bf16(
          /*neg_a=*/false, a, /*neg_b=*/false, bm,
          /*c_mod=*/(short)0, c, /*reuse_a=*/false, /*reuse_b=*/false);
    }
    // epilogue: C/D layout — VGPR r, lane: M = r + half*8, N = lane&15
#pragma unroll
    for (int r = 0; r < 8; ++r) {
      int co = ct * 16 + r + half * 8;
      if (co < COUT) {
        float v = c[r] + bias[co];
        long long off = (((long long)b * COUT + co) * H + y) * W + (x0 + px);
        out[off] = v;
        atomicAdd(&sstat[co * 2 + 0], v);
        atomicAdd(&sstat[co * 2 + 1], v * v);
      }
    }
  }
  __syncthreads();
  for (int i = tid; i < COUT * 2; i += 256) atomicAdd(&stats[i], sstat[i]);
}

// ---------- 4) BN parameter fold: scale = g*rsqrt(var+eps), shift = be-mean*scale
__global__ void bn_param_kernel(const float* __restrict__ stats,
                                const float* __restrict__ g,
                                const float* __restrict__ be,
                                float* __restrict__ scale,
                                float* __restrict__ shift,
                                int C, float invN) {
  int c = threadIdx.x;
  if (c >= C) return;
  float mean = stats[c * 2 + 0] * invN;
  float var  = stats[c * 2 + 1] * invN - mean * mean;
  float s    = g[c] * rsqrtf(var + 1e-5f);
  scale[c] = s;
  shift[c] = be[c] - mean * s;
}

// ---------- 5) fused BN + ReLU (+ optional 2x2 maxpool) ----------
template<bool POOL>
__global__ void bn_act_kernel(const float* __restrict__ in, float* __restrict__ out,
                              const float* __restrict__ scale,
                              const float* __restrict__ shift,
                              int C, int H, int W, long long total) {
  long long i = (long long)blockIdx.x * blockDim.x + threadIdx.x;
  if (i >= total) return;
  int OW = POOL ? (W >> 1) : W;
  int OH = POOL ? (H >> 1) : H;
  int x = (int)(i % OW); long long t = i / OW;
  int y = (int)(t % OH); t /= OH;
  int c = (int)(t % C);  long long b = t / C;
  float sc = scale[c], sh = shift[c];
  long long rowbase = ((b * C + c) * (long long)H);
  if (POOL) {
    const float* p = in + (rowbase + y * 2) * W + x * 2;
    float v00 = fmaf(p[0],     sc, sh);
    float v01 = fmaf(p[1],     sc, sh);
    float v10 = fmaf(p[W],     sc, sh);
    float v11 = fmaf(p[W + 1], sc, sh);
    out[i] = fmaxf(fmaxf(fmaxf(v00, v01), fmaxf(v10, v11)), 0.0f);
  } else {
    out[i] = fmaxf(fmaf(in[(rowbase + y) * W + x], sc, sh), 0.0f);
  }
}

// ---------- launch ----------
extern "C" void kernel_launch(void* const* d_in, const int* in_sizes, int n_in,
                              void* d_out, int out_size, void* d_ws, size_t ws_size,
                              hipStream_t stream) {
  (void)n_in; (void)out_size; (void)ws_size;
  const int B = 4;
  const float* pts = (const float*)d_in[0];
  const int N = in_sizes[0] / 5;
  const float *w1 = (const float*)d_in[1],  *b1 = (const float*)d_in[2];
  const float *g1 = (const float*)d_in[3],  *be1 = (const float*)d_in[4];
  const float *w2 = (const float*)d_in[5],  *b2 = (const float*)d_in[6];
  const float *g2 = (const float*)d_in[7],  *be2 = (const float*)d_in[8];
  const float *w3 = (const float*)d_in[9],  *b3 = (const float*)d_in[10];
  const float *g3 = (const float*)d_in[11], *be3 = (const float*)d_in[12];
  const float *w4 = (const float*)d_in[13], *b4 = (const float*)d_in[14];
  const float *g4 = (const float*)d_in[15], *be4 = (const float*)d_in[16];

  // workspace layout
  float* wsf = (float*)d_ws;
  float* stats0 = wsf + 0 * 64;  float* stats1 = wsf + 1 * 64;
  float* stats2 = wsf + 2 * 64;  float* stats3 = wsf + 3 * 64;
  float* scl0 = wsf + 256 + 0 * 64; float* scl1 = wsf + 256 + 1 * 64;
  float* scl2 = wsf + 256 + 2 * 64; float* scl3 = wsf + 256 + 3 * 64;
  float* sh0  = wsf + 512 + 0 * 64; float* sh1  = wsf + 512 + 1 * 64;
  float* sh2  = wsf + 512 + 2 * 64; float* sh3  = wsf + 512 + 3 * 64;

  char* base = (char*)d_ws;
  const size_t MB = 1ull << 20;
  char* regA = base + 1   * MB;   // 32 MB:  ubev -> p1 -> t3
  char* regB = base + 33  * MB;   // 128 MB: t1 -> t2 -> a3
  char* regC = base + 161 * MB;   // 32 MB:  p2 -> t4

  unsigned* ubev = (unsigned*)regA;
  float* t1 = (float*)regB;
  float* p1 = (float*)regA;
  float* t2 = (float*)regB;
  float* p2 = (float*)regC;
  float* t3 = (float*)regA;
  float* a3 = (float*)regB;
  float* t4 = (float*)regC;
  float* outp = (float*)d_out;

  const long long nbev = (long long)B * 2 * 1024 * 1024;
  init_kernel<<<(unsigned)((nbev + 255) / 256), 256, 0, stream>>>(ubev, wsf, nbev);
  scatter_kernel<<<(N + 255) / 256, 256, 0, stream>>>(pts, ubev, N);

  // layer 1: 2 -> 8, 1024x1024, pool -> 512
  conv_wmma_kernel<2, 8, true><<<dim3(8, 1024, B), 256, 0, stream>>>(
      ubev, w1, b1, t1, stats0, 1024, 1024);
  bn_param_kernel<<<1, 64, 0, stream>>>(stats0, g1, be1, scl0, sh0, 8,
                                        1.0f / (float)(B * 1024 * 1024));
  {
    long long tot = (long long)B * 8 * 512 * 512;
    bn_act_kernel<true><<<(unsigned)((tot + 255) / 256), 256, 0, stream>>>(
        t1, p1, scl0, sh0, 8, 1024, 1024, tot);
  }

  // layer 2: 8 -> 16, 512x512, pool -> 256
  conv_wmma_kernel<8, 16, false><<<dim3(4, 512, B), 256, 0, stream>>>(
      p1, w2, b2, t2, stats1, 512, 512);
  bn_param_kernel<<<1, 64, 0, stream>>>(stats1, g2, be2, scl1, sh1, 16,
                                        1.0f / (float)(B * 512 * 512));
  {
    long long tot = (long long)B * 16 * 256 * 256;
    bn_act_kernel<true><<<(unsigned)((tot + 255) / 256), 256, 0, stream>>>(
        t2, p2, scl1, sh1, 16, 512, 512, tot);
  }

  // layer 3: 16 -> 32, 256x256, no pool
  conv_wmma_kernel<16, 32, false><<<dim3(2, 256, B), 256, 0, stream>>>(
      p2, w3, b3, t3, stats2, 256, 256);
  bn_param_kernel<<<1, 64, 0, stream>>>(stats2, g3, be3, scl2, sh2, 32,
                                        1.0f / (float)(B * 256 * 256));
  {
    long long tot = (long long)B * 32 * 256 * 256;
    bn_act_kernel<false><<<(unsigned)((tot + 255) / 256), 256, 0, stream>>>(
        t3, a3, scl2, sh2, 32, 256, 256, tot);
  }

  // layer 4: 32 -> 32, 256x256, pool -> 128 (final output)
  conv_wmma_kernel<32, 32, false><<<dim3(2, 256, B), 256, 0, stream>>>(
      a3, w4, b4, t4, stats3, 256, 256);
  bn_param_kernel<<<1, 64, 0, stream>>>(stats3, g4, be4, scl3, sh3, 32,
                                        1.0f / (float)(B * 256 * 256));
  {
    long long tot = (long long)B * 32 * 128 * 128;
    bn_act_kernel<true><<<(unsigned)((tot + 255) / 256), 256, 0, stream>>>(
        t4, outp, scl3, sh3, 32, 256, 256, tot);
  }
}
